// RUMModel_52819507806214
// MI455X (gfx1250) — matrix operand order repaired and checked by reading, not compiled, not run on hardware
//
#include <hip/hip_runtime.h>

// ---------------------------------------------------------------------------
// RUM model for MI455X (gfx1250, wave32, WMMA 16x16x32 f16 -> f32 accum).
// Sizes (fixed by the reference): B=8192, n=50, dx=32, dz=20,
// WX=WZ=64, WU=128, LX=LZ=16, KX=KZ=3, depths 3, TOL=1e-6.
// ---------------------------------------------------------------------------

typedef __attribute__((ext_vector_type(16))) _Float16 v16h;
typedef __attribute__((ext_vector_type(8)))  float    v8f;

#define DEVINL static __device__ __forceinline__

#define BATCH   8192
#define NPROD   50
#define BN      409600          // BATCH * NPROD
#define TILES   25600           // BN / 16
#define ZTILES  512             // BATCH / 16

// ws layout in halves (f16). Weights first, then staged activations.
#define OFF_WX0   0             // [3][32][64]
#define OFF_WX    6144          // [3][2][64][64]
#define OFF_WLX   30720         // [64][16]      (only Wlx[2] -- faithful bug)
#define OFF_WZ0   31744         // [3][32][64]   (dz=20 zero-padded to 32)
#define OFF_WZ    37888         // [3][2][64][64]
#define OFF_WLZ   62464         // [3][64][16]
#define OFF_WU0   65536         // [96][128]     (84 zero-padded to 96)
#define OFF_WU    77824         // [2][128][128]
#define OFF_ZJ    110592        // [3][8192][16] f16
#define OFF_XI    503808        // [3][409600][16] f16
#define OFF_END   20164608      // halves; scores (f32 [9][409600]) follow

struct Frag {
  union { v16h v; uint4 q[2]; };
};

DEVINL int lane_id() { return (int)(threadIdx.x & 31u); }

// Wave-local DS store->load ordering (split counters on CDNA5).
DEVINL void lds_fence() { asm volatile("s_wait_dscnt 0" ::: "memory"); }

// CDNA5 async global->LDS copy (bypasses VGPRs; tracked by ASYNCcnt).
// VDST operand is the per-lane LDS byte address; low 32 bits of a generic
// pointer to __shared__ are the LDS offset on AMDGPU.
DEVINL void async_copy_b128(unsigned lds_off, const void* gptr) {
  asm volatile("global_load_async_to_lds_b128 %0, %1, off"
               :: "v"(lds_off), "v"(gptr) : "memory");
}
DEVINL void async_wait0() { asm volatile("s_wait_asynccnt 0" ::: "memory"); }

// A fragment (16 rows x 32 K) from row-major f16 [16 x pitch] at column kBase.
// Lane l (M = l%16) holds K-chunks [kc..kc+7] and [kc+16..kc+23], kc = (l<16?0:8).
DEVINL Frag load_a_f16(const _Float16* src, int pitch, int kBase) {
  int m  = (int)(threadIdx.x & 15u);
  int kc = kBase + ((threadIdx.x & 16u) ? 8 : 0);
  const _Float16* p = src + m * pitch + kc;
  Frag f;
  f.q[0] = *(const uint4*)(p);
  f.q[1] = *(const uint4*)(p + 16);
  return f;
}

// A fragment from row-major f32 [16 x pitch]; columns >= KV read as zero.
template <int KV>
DEVINL Frag load_a_f32(const float* src, int pitch) {
  int m  = (int)(threadIdx.x & 15u);
  int kc = (threadIdx.x & 16u) ? 8 : 0;
  const float* p = src + m * pitch;
  Frag f;
#pragma unroll
  for (int h = 0; h < 8; ++h) {
    int k0 = kc + h, k1 = kc + 16 + h;
    f.v[h]     = (_Float16)((k0 < KV) ? p[k0] : 0.0f);
    f.v[h + 8] = (_Float16)((k1 < KV) ? p[k1] : 0.0f);
  }
  return f;
}

// B fragment (32 K x 16 N) from row-major f16 weights [K x pitch] at (kBase,nBase).
// Lane l holds column N = nBase + l%16, K-chunks [kc..kc+7] and [kc+16..kc+23], kc = kBase + (l<16?0:8).
DEVINL Frag load_b(const _Float16* w, int pitch, int kBase, int nBase) {
  int n  = nBase + (int)(threadIdx.x & 15u);
  int kc = kBase + ((threadIdx.x & 16u) ? 8 : 0);
  Frag f;
#pragma unroll
  for (int s = 0; s < 8; ++s) {
    f.v[s]     = w[(kc + s) * pitch + n];
    f.v[s + 8] = w[(kc + 16 + s) * pitch + n];
  }
  return f;
}

DEVINL v8f wmma16(const Frag& a, const Frag& b, v8f c) {
  return __builtin_amdgcn_wmma_f32_16x16x32_f16(false, a.v, false, b.v,
                                                (short)0, c, false, false);
}

DEVINL float elu(float x) { return x > 0.0f ? x : (__expf(x) - 1.0f); }

// C tile (f32, lane l: M = r + (l<16?0:8), N = nBase + l%16) -> elu -> f16 LDS.
DEVINL void elu_store(v8f c, _Float16* dst, int pitch, int nBase) {
  int l    = lane_id();
  int mOff = (l & 16) ? 8 : 0;
  int col  = nBase + (l & 15);
#pragma unroll
  for (int r = 0; r < 8; ++r)
    dst[(mOff + r) * pitch + col] = (_Float16)elu(c[r]);
}

// ---------------------------------------------------------------------------
// Kernel 1: convert all weights to f16 in ws (with zero padding where needed).
// ---------------------------------------------------------------------------
__global__ __launch_bounds__(256) void prep_weights_kernel(
    const float* __restrict__ Wx0, const float* __restrict__ Wx,
    const float* __restrict__ Wlx, const float* __restrict__ Wz0,
    const float* __restrict__ Wz,  const float* __restrict__ Wlz,
    const float* __restrict__ Wu0, const float* __restrict__ Wu,
    _Float16* __restrict__ ws) {
  int idx = blockIdx.x * 256 + (int)threadIdx.x;
  float v;
  if (idx < 6144)        v = Wx0[idx];
  else if (idx < 30720)  v = Wx[idx - 6144];
  else if (idx < 31744)  v = Wlx[2 * 64 * 16 + (idx - 30720)];   // Wlx[DX-1] only
  else if (idx < 37888) {
    int t = idx - 31744; int n = t & 63; int k = (t >> 6) & 31; int j = t >> 11;
    v = (k < 20) ? Wz0[(j * 20 + k) * 64 + n] : 0.0f;
  } else if (idx < 62464) v = Wz[idx - 37888];
  else if (idx < 65536)   v = Wlz[idx - 62464];
  else if (idx < 77824) {
    int t = idx - 65536; int n = t & 127; int k = t >> 7;
    v = (k < 84) ? Wu0[k * 128 + n] : 0.0f;
  } else if (idx < 110592) v = Wu[idx - 77824];
  else return;
  ws[idx] = (_Float16)v;
}

// ---------------------------------------------------------------------------
// Kernel 2: z_branch. One wave per 16-row tile. Zj[j][b][16] in f16.
// ---------------------------------------------------------------------------
__global__ __launch_bounds__(32) void z_branch_kernel(
    const float* __restrict__ Z, const _Float16* __restrict__ ws,
    _Float16* __restrict__ Zj) {
  __shared__ __align__(16) _Float16 act[16 * 64];
  int j = blockIdx.x >> 9;            // / ZTILES
  int tile = blockIdx.x & (ZTILES - 1);
  int rowBase = tile << 4;
  const _Float16* W0 = ws + OFF_WZ0 + j * 2048;   // [32][64]
  const _Float16* Wh = ws + OFF_WZ  + j * 8192;   // [2][64][64]
  const _Float16* Wl = ws + OFF_WLZ + j * 1024;   // [64][16]

  Frag a0 = load_a_f32<20>(Z + (size_t)rowBase * 20, 20);
#pragma unroll
  for (int nt = 0; nt < 4; ++nt) {
    v8f z = {};
    z = wmma16(a0, load_b(W0, 64, 0, nt * 16), z);
    elu_store(z, act, 64, nt * 16);
  }
  lds_fence();
#pragma unroll
  for (int L = 0; L < 2; ++L) {
    Frag a1 = load_a_f16(act, 64, 0);
    Frag a2 = load_a_f16(act, 64, 32);
    v8f cc[4];
#pragma unroll
    for (int nt = 0; nt < 4; ++nt) {
      v8f z = {};
      z = wmma16(a1, load_b(Wh + L * 4096, 64, 0,  nt * 16), z);
      z = wmma16(a2, load_b(Wh + L * 4096, 64, 32, nt * 16), z);
      cc[nt] = z;
    }
#pragma unroll
    for (int nt = 0; nt < 4; ++nt) elu_store(cc[nt], act, 64, nt * 16);
    lds_fence();
  }
  Frag a1 = load_a_f16(act, 64, 0);
  Frag a2 = load_a_f16(act, 64, 32);
  v8f s = {};
  s = wmma16(a1, load_b(Wl, 16, 0, 0), s);
  s = wmma16(a2, load_b(Wl, 16, 32, 0), s);
  int l = lane_id(); int mOff = (l & 16) ? 8 : 0; int col = l & 15;
#pragma unroll
  for (int r = 0; r < 8; ++r)
    Zj[((size_t)j * BATCH + rowBase + mOff + r) * 16 + col] = (_Float16)s[r];
}

// ---------------------------------------------------------------------------
// Kernel 3: x_branch. 8 waves/block, weights for one i async-staged in LDS.
// Xi[i][t][16] f16; last layer uses Wlx[2] for every i (faithful bug).
// ---------------------------------------------------------------------------
__global__ __launch_bounds__(256) void x_branch_kernel(
    const float* __restrict__ X, const _Float16* __restrict__ ws,
    _Float16* __restrict__ Xi) {
  __shared__ __align__(16) _Float16 sW[11264];        // 22KB
  __shared__ __align__(16) _Float16 act[8][16 * 64];  // 16KB
  int i   = blockIdx.x / 1200;
  int blk = blockIdx.x % 1200;
  {
    unsigned base = (unsigned)(size_t)(&sW[0]);
    for (int t = (int)threadIdx.x; t < 1408; t += 256) {
      const char* src;
      if (t < 256)       src = (const char*)(ws + OFF_WX0 + i * 2048) + (size_t)t * 16;
      else if (t < 1280) src = (const char*)(ws + OFF_WX  + i * 8192) + (size_t)(t - 256) * 16;
      else               src = (const char*)(ws + OFF_WLX)            + (size_t)(t - 1280) * 16;
      async_copy_b128(base + (unsigned)t * 16u, src);
    }
    async_wait0();
  }
  __syncthreads();
  int wave = (int)(threadIdx.x >> 5);
  const _Float16* W0 = sW;            // [32][64]
  const _Float16* Wh = sW + 2048;     // [2][64][64]
  const _Float16* Wl = sW + 10240;    // [64][16]
  _Float16* A = act[wave];

  for (int tile = blk * 8 + wave; tile < TILES; tile += 9600) {
    int rowBase = tile << 4;
    int nxt = tile + 9600;
    if (nxt < TILES) __builtin_prefetch(X + (size_t)nxt * 16 * 32, 0, 1);

    Frag a0 = load_a_f32<32>(X + (size_t)rowBase * 32, 32);
#pragma unroll
    for (int nt = 0; nt < 4; ++nt) {
      v8f z = {};
      z = wmma16(a0, load_b(W0, 64, 0, nt * 16), z);
      elu_store(z, A, 64, nt * 16);
    }
    lds_fence();
#pragma unroll
    for (int L = 0; L < 2; ++L) {
      Frag a1 = load_a_f16(A, 64, 0);
      Frag a2 = load_a_f16(A, 64, 32);
      v8f cc[4];
#pragma unroll
      for (int nt = 0; nt < 4; ++nt) {
        v8f z = {};
        z = wmma16(a1, load_b(Wh + L * 4096, 64, 0,  nt * 16), z);
        z = wmma16(a2, load_b(Wh + L * 4096, 64, 32, nt * 16), z);
        cc[nt] = z;
      }
#pragma unroll
      for (int nt = 0; nt < 4; ++nt) elu_store(cc[nt], A, 64, nt * 16);
      lds_fence();
    }
    Frag a1 = load_a_f16(A, 64, 0);
    Frag a2 = load_a_f16(A, 64, 32);
    v8f s = {};
    s = wmma16(a1, load_b(Wl, 16, 0, 0), s);
    s = wmma16(a2, load_b(Wl, 16, 32, 0), s);
    int l = lane_id(); int mOff = (l & 16) ? 8 : 0; int col = l & 15;
#pragma unroll
    for (int r = 0; r < 8; ++r)
      Xi[((size_t)i * BN + rowBase + mOff + r) * 16 + col] = (_Float16)s[r];
    lds_fence();
  }
}

// ---------------------------------------------------------------------------
// Kernel 4: utility MLP over all 9 (i,j) pairs. 88 WMMA per 16-token tile.
// ALL utility weights (Wu0 + Wu[0] + Wu[1], 90KB) async-staged once and kept
// resident in LDS (CDNA5 WGP has 320KB): no per-layer restaging, no in-loop
// barriers, waves fully independent. Per-wave 4KB activation buffer.
// Final 128->1 layer as in-register dot + 16-lane shfl_xor reduction.
// ---------------------------------------------------------------------------
#define U_GRID 1152
__global__ __launch_bounds__(256) void util_kernel(
    const float* __restrict__ X, const float* __restrict__ Z,
    const float* __restrict__ Wlast, const _Float16* __restrict__ ws,
    const _Float16* __restrict__ Xi, const _Float16* __restrict__ Zj,
    float* __restrict__ scores) {
  __shared__ __align__(16) _Float16 sW[45056];         // 88KB: Wu0 | Wu0h | Wu1h
  __shared__ __align__(16) _Float16 act[8][16 * 128];  // 32KB: per-wave activ.
  int wave = (int)(threadIdx.x >> 5);
  int l = lane_id();
  int m = l & 15;
  int hiK = (l >> 4) & 1;
  float wl[8];
#pragma unroll
  for (int nt = 0; nt < 8; ++nt) wl[nt] = Wlast[nt * 16 + m];
  _Float16* A = act[wave];

  // One-time async staging of all 45056 weight halves (contiguous in ws).
  {
    const char* gW = (const char*)(ws + OFF_WU0);
    unsigned base = (unsigned)(size_t)(&sW[0]);
    for (int t = (int)threadIdx.x; t < 5632; t += 256)
      async_copy_b128(base + (unsigned)t * 16u, gW + (size_t)t * 16);
    async_wait0();
  }
  __syncthreads();
  const _Float16* W0 = sW;            // [96][128]
  const _Float16* Wh = sW + 12288;    // [2][128][128]

  for (int job = blockIdx.x * 8 + wave; job < 9 * TILES; job += U_GRID * 8) {
    int p = job / TILES;
    int tile = job % TILES;
    int i = p / 3, jj = p % 3;
    int rowBase = tile << 4;
    int t = rowBase + m;
    int b = t / NPROD;

    __builtin_prefetch(Xi + ((size_t)i * BN + rowBase) * 16, 0, 1);

    // u = [X(0..31) | Xi(32..47) | Zj(48..63) | Z(64..83) | pad(84..95)]
    Frag a0 = load_a_f32<32>(X + (size_t)rowBase * 32, 32);
    Frag a1;
    a1.q[0] = *(const uint4*)(Xi + ((size_t)i * BN + t) * 16 + hiK * 8);
    a1.q[1] = *(const uint4*)(Zj + ((size_t)jj * BATCH + b) * 16 + hiK * 8);
    Frag a2;
    {
      const float* Zr = Z + (size_t)b * 20;
#pragma unroll
      for (int h = 0; h < 8; ++h) {
        int k0 = hiK * 8 + h, k1 = k0 + 16;
        a2.v[h]     = (_Float16)((k0 < 20) ? Zr[k0] : 0.0f);
        a2.v[h + 8] = (_Float16)((k1 < 20) ? Zr[k1] : 0.0f);
      }
    }
    v8f c[8];
#pragma unroll
    for (int nt = 0; nt < 8; ++nt) {                   // layer 0: 96 -> 128
      v8f z = {};
      z = wmma16(a0, load_b(W0, 128, 0,  nt * 16), z);
      z = wmma16(a1, load_b(W0, 128, 32, nt * 16), z);
      z = wmma16(a2, load_b(W0, 128, 64, nt * 16), z);
      c[nt] = z;
    }
#pragma unroll
    for (int nt = 0; nt < 8; ++nt) elu_store(c[nt], A, 128, nt * 16);
    lds_fence();

#pragma unroll
    for (int L = 0; L < 2; ++L) {                      // hidden: 128 -> 128
      Frag a[4];
#pragma unroll
      for (int ks = 0; ks < 4; ++ks) a[ks] = load_a_f16(A, 128, ks * 32);
#pragma unroll
      for (int nt = 0; nt < 8; ++nt) {
        v8f z = {};
#pragma unroll
        for (int ks = 0; ks < 4; ++ks)
          z = wmma16(a[ks], load_b(Wh + L * 16384, 128, ks * 32, nt * 16), z);
        c[nt] = z;
      }
      if (L == 0) {
#pragma unroll
        for (int nt = 0; nt < 8; ++nt) elu_store(c[nt], A, 128, nt * 16);
        lds_fence();
      }
    }

    // last layer: score = elu(c) . Wlast (f32, in registers), reduce over N lanes
    float s[8];
#pragma unroll
    for (int r = 0; r < 8; ++r) {
      float acc = 0.0f;
#pragma unroll
      for (int nt = 0; nt < 8; ++nt) acc += elu(c[nt][r]) * wl[nt];
      s[r] = acc;
    }
#pragma unroll
    for (int d = 1; d < 16; d <<= 1) {
#pragma unroll
      for (int r = 0; r < 8; ++r) s[r] += __shfl_xor(s[r], d, 32);
    }
    if (m == 0) {
      int mOff = hiK * 8;
#pragma unroll
      for (int r = 0; r < 8; ++r)
        scores[(size_t)p * BN + rowBase + mOff + r] = s[r];
    }
  }
}

// ---------------------------------------------------------------------------
// Kernel 5: per-(pair,b) softmax over n=50, TOL smoothing, average of 9 pairs.
// ---------------------------------------------------------------------------
__global__ __launch_bounds__(64) void softmax_acc_kernel(
    const float* __restrict__ scores, float* __restrict__ out) {
  __shared__ float red[64];
  int b = blockIdx.x;
  int t = (int)threadIdx.x;
  const float TOLc = 1e-6f;
  float acc = 0.0f;
  for (int p = 0; p < 9; ++p) {
    float sc = (t < NPROD) ? scores[(size_t)p * BN + (size_t)b * NPROD + t]
                           : -3.4e38f;
    red[t] = sc; __syncthreads();
#pragma unroll
    for (int d = 32; d > 0; d >>= 1) {
      if (t < d) red[t] = fmaxf(red[t], red[t + d]);
      __syncthreads();
    }
    float mx = red[0]; __syncthreads();
    float e = (t < NPROD) ? __expf(sc - mx) : 0.0f;
    red[t] = e; __syncthreads();
#pragma unroll
    for (int d = 32; d > 0; d >>= 1) {
      if (t < d) red[t] += red[t + d];
      __syncthreads();
    }
    float sum = red[0]; __syncthreads();
    if (t < NPROD) acc += (TOLc + e / sum) / (1.0f + TOLc * (float)NPROD);
  }
  if (t < NPROD) out[(size_t)b * NPROD + t] = acc * (1.0f / 9.0f);
}

// ---------------------------------------------------------------------------
extern "C" void kernel_launch(void* const* d_in, const int* in_sizes, int n_in,
                              void* d_out, int out_size, void* d_ws, size_t ws_size,
                              hipStream_t stream) {
  (void)in_sizes; (void)n_in; (void)out_size; (void)ws_size;
  const float* X     = (const float*)d_in[0];
  const float* Z     = (const float*)d_in[1];
  const float* Wx0   = (const float*)d_in[2];
  const float* Wx    = (const float*)d_in[3];
  const float* Wlx   = (const float*)d_in[4];
  const float* Wz0   = (const float*)d_in[5];
  const float* Wz    = (const float*)d_in[6];
  const float* Wlz   = (const float*)d_in[7];
  const float* Wu0   = (const float*)d_in[8];
  const float* Wu    = (const float*)d_in[9];
  const float* Wlast = (const float*)d_in[10];
  float* out = (float*)d_out;

  _Float16* wsh    = (_Float16*)d_ws;            // ~55 MB total used
  _Float16* Zj     = wsh + OFF_ZJ;
  _Float16* Xi     = wsh + OFF_XI;
  float*    scores = (float*)(wsh + OFF_END);

  prep_weights_kernel<<<432, 256, 0, stream>>>(Wx0, Wx, Wlx, Wz0, Wz, Wlz,
                                               Wu0, Wu, wsh);
  z_branch_kernel<<<3 * ZTILES, 32, 0, stream>>>(Z, wsh, Zj);
  x_branch_kernel<<<3 * 1200, 256, 0, stream>>>(X, wsh, Xi);
  util_kernel<<<U_GRID, 256, 0, stream>>>(X, Z, Wlast, wsh, Xi, Zj, scores);
  softmax_acc_kernel<<<BATCH, 64, 0, stream>>>(scores, out);
}
